// CrossAttentionBlock_11476152615208
// MI455X (gfx1250) — matrix operand order repaired and checked by reading, not compiled, not run on hardware
//
#include <hip/hip_runtime.h>
#include <math.h>

// MI455X / gfx1250: wave32, WMMA f32_16x16x32_f16 + double-buffered TDM staging.

typedef __attribute__((ext_vector_type(16))) _Float16 v16h;
typedef __attribute__((ext_vector_type(8)))  _Float16 v8h;
typedef __attribute__((ext_vector_type(8)))  float    v8f;
typedef __attribute__((ext_vector_type(4)))  unsigned int v4u;
typedef __attribute__((ext_vector_type(8)))  int v8i;
typedef __attribute__((ext_vector_type(4)))  int v4i;

#define HDIM 768
#define DHEAD 64
#define EPSF 1e-5f
#define SCALEF 0.125f   // 1/sqrt(64), exact in f16

__device__ __forceinline__ _Float16 f2h(float f) { return (_Float16)f; }

union HFrag { v16h v; v8h h[2]; };

__device__ __forceinline__ v8f wmma16(const HFrag& a, const HFrag& b, v8f c) {
  return __builtin_amdgcn_wmma_f32_16x16x32_f16(false, a.v, false, b.v,
                                                (short)0, c, false, false);
}

__device__ __forceinline__ v8f vzero8() {
  v8f z = {0.f, 0.f, 0.f, 0.f, 0.f, 0.f, 0.f, 0.f};
  return z;
}

// ---------------------------------------------------------------------------
// TDM: load a 128-row x 32-elem f16 tile (row stride = rowLen elems) from
// global into LDS at lds_off, padding 16B after every 64B row -> LDS row
// stride of 40 halves (bank-conflict-free fragment reads).
// D# per CDNA5 ISA section 8.
// ---------------------------------------------------------------------------
__device__ __forceinline__ void tdm_load_tile_f16(unsigned lds_off,
                                                  const _Float16* gptr,
                                                  unsigned rowLen,
                                                  unsigned nRows) {
  unsigned long long ga = (unsigned long long)(uintptr_t)gptr;
  v4u g0;
  g0[0] = 1u;                                  // count=1 (valid user D#)
  g0[1] = lds_off;                             // LDS byte address
  g0[2] = (unsigned)ga;                        // global addr [31:0]
  g0[3] = (unsigned)((ga >> 32) & 0x1FFFFFFu)  // global addr [56:32]
          | (2u << 30);                        // type = 2 ("image")
  v8i g1;
  // data_size=1 (2B), pad_enable, pad_interval=3 (16 DWORDs), pad_amount=3 (4 DWORDs)
  g1[0] = (int)((1u << 16) | (1u << 20) | (3u << 22) | (3u << 25));
  g1[1] = (int)((rowLen & 0xFFFFu) << 16);                       // dim0[15:0]
  g1[2] = (int)(((rowLen >> 16) & 0xFFFFu) | ((nRows & 0xFFFFu) << 16));
  g1[3] = (int)(((nRows >> 16) & 0xFFFFu) | (32u << 16));        // tile_dim0=32
  g1[4] = (int)128;                                              // tile_dim1=128
  g1[5] = (int)rowLen;                                           // dim0 stride lo
  g1[6] = 0;
  g1[7] = 0;
  v4i z4 = {0, 0, 0, 0};
#if __clang_major__ >= 23
  v8i z8 = {0, 0, 0, 0, 0, 0, 0, 0};
  __builtin_amdgcn_tensor_load_to_lds(g0, g1, z4, z4, z8, 0);
#else
  __builtin_amdgcn_tensor_load_to_lds(g0, g1, z4, z4, 0);
#endif
}

// ---------------------------------------------------------------------------
// GEMM: Y[M,N](f32) = act(A[M,K](f16) @ Wt[N,K](f16)^T + bias)
// Block 256 = 8 waves; tile 128x128x32; wave computes 64x32 (8 accumulators).
// Double-buffered TDM staging: DMA of tile kb+1 overlaps WMMA of tile kb.
// TENSORcnt is in-order per wave, so waiting <=2 after issuing the next pair
// guarantees the current pair has landed.
// ---------------------------------------------------------------------------
template <int ACT>
__global__ void __launch_bounds__(256)
gemm_h_kernel(const _Float16* __restrict__ A, const _Float16* __restrict__ Wt,
              const float* __restrict__ bias, float* __restrict__ Y,
              int M, int K, int N) {
  __shared__ __align__(16) _Float16 As[2][128][40];  // ping-pong, TDM-padded
  __shared__ __align__(16) _Float16 Bs[2][128][40];

  const int tid = threadIdx.x;
  const int lane = tid & 31, wid = tid >> 5;
  const int waveM = wid >> 2, waveN = wid & 3;
  const int rowBase = blockIdx.y * 128, colBase = blockIdx.x * 128;
  const unsigned ldsA[2] = {(unsigned)(uintptr_t)&As[0][0][0],
                            (unsigned)(uintptr_t)&As[1][0][0]};
  const unsigned ldsB[2] = {(unsigned)(uintptr_t)&Bs[0][0][0],
                            (unsigned)(uintptr_t)&Bs[1][0][0]};
  const _Float16* Abase = A + (size_t)rowBase * K;
  const _Float16* Bbase = Wt + (size_t)colBase * K;

  v8f acc[4][2];
#pragma unroll
  for (int i = 0; i < 4; ++i)
#pragma unroll
    for (int j = 0; j < 2; ++j) acc[i][j] = vzero8();

  const int kTiles = K >> 5;
  if (wid == 0) {  // prologue: stage tile 0 into buffer 0
    tdm_load_tile_f16(ldsA[0], Abase, (unsigned)K, (unsigned)M);
    tdm_load_tile_f16(ldsB[0], Bbase, (unsigned)K, (unsigned)N);
  }

  for (int kb = 0; kb < kTiles; ++kb) {
    const int cur = kb & 1;
    if (wid == 0) {
      if (kb + 1 < kTiles) {  // prefetch next tile into the idle buffer
        tdm_load_tile_f16(ldsA[cur ^ 1], Abase + ((kb + 1) << 5),
                          (unsigned)K, (unsigned)M);
        tdm_load_tile_f16(ldsB[cur ^ 1], Bbase + ((kb + 1) << 5),
                          (unsigned)K, (unsigned)N);
        __builtin_amdgcn_s_wait_tensorcnt(2);  // current tile landed
      } else {
        __builtin_amdgcn_s_wait_tensorcnt(0);  // last tile landed
      }
    }
    __syncthreads();

    HFrag afr[4], bfr[2];
    const int ar = lane & 15;
    const int ak = (lane >> 4) << 3;   // 0 or 8
#pragma unroll
    for (int mt = 0; mt < 4; ++mt) {
      int rr = waveM * 64 + mt * 16 + ar;
      afr[mt].h[0] = *(const v8h*)&As[cur][rr][ak];
      afr[mt].h[1] = *(const v8h*)&As[cur][rr][ak + 16];
    }
    const int bn = lane & 15;
    const int bk = (lane >> 4) << 3;   // 0 or 8
#pragma unroll
    for (int nt = 0; nt < 2; ++nt) {
      int nn = waveN * 32 + nt * 16 + bn;
      bfr[nt].h[0] = *(const v8h*)&Bs[cur][nn][bk];
      bfr[nt].h[1] = *(const v8h*)&Bs[cur][nn][bk + 16];
    }
#pragma unroll
    for (int mt = 0; mt < 4; ++mt)
#pragma unroll
      for (int nt = 0; nt < 2; ++nt)
        acc[mt][nt] = wmma16(afr[mt], bfr[nt], acc[mt][nt]);
    __syncthreads();  // all waves done with buf[cur] before it is re-filled
  }

#pragma unroll
  for (int mt = 0; mt < 4; ++mt) {
#pragma unroll
    for (int nt = 0; nt < 2; ++nt) {
      int col = colBase + waveN * 32 + nt * 16 + (lane & 15);
      float bv = bias ? bias[col] : 0.f;
#pragma unroll
      for (int r = 0; r < 8; ++r) {
        int row = rowBase + waveM * 64 + mt * 16 + ((lane >> 4) << 3) + r;
        float x = acc[mt][nt][r] + bv;
        if (ACT == 1) x = 0.5f * x * (1.f + erff(x * 0.70710678118f));
        Y[(size_t)row * N + col] = x;
      }
    }
  }
}

// ---------------------------------------------------------------------------
// f32 -> f16 bulk convert, 8 elems/thread, 16B stores.
// ---------------------------------------------------------------------------
__global__ void __launch_bounds__(256)
cvt_kernel(const float* __restrict__ src, _Float16* __restrict__ dst) {
  size_t i = ((size_t)blockIdx.x * 256 + threadIdx.x) * 8;
  const float4 a = *(const float4*)(src + i);
  const float4 b = *(const float4*)(src + i + 4);
  v8h o;
  o[0] = f2h(a.x); o[1] = f2h(a.y); o[2] = f2h(a.z); o[3] = f2h(a.w);
  o[4] = f2h(b.x); o[5] = f2h(b.y); o[6] = f2h(b.z); o[7] = f2h(b.w);
  *(v8h*)(dst + i) = o;
}

// ---------------------------------------------------------------------------
// Weight transpose + convert: Wt[N][K](f16) = W[K][N](f32)^T. 32x32 LDS tiles.
// ---------------------------------------------------------------------------
__global__ void __launch_bounds__(256)
twcvt_kernel(const float* __restrict__ W, _Float16* __restrict__ Wt,
             int K, int N) {
  __shared__ float tile[32][33];
  int kb = blockIdx.y * 32, nb = blockIdx.x * 32;
  int tid = threadIdx.x;
#pragma unroll
  for (int i = 0; i < 4; ++i) {
    int idx = tid + i * 256;
    int r = idx >> 5, c = idx & 31;
    tile[r][c] = W[(size_t)(kb + r) * N + nb + c];
  }
  __syncthreads();
#pragma unroll
  for (int i = 0; i < 4; ++i) {
    int idx = tid + i * 256;
    int r = idx >> 5, c = idx & 31;
    Wt[(size_t)(nb + r) * K + kb + c] = f2h(tile[c][r]);
  }
}

// ---------------------------------------------------------------------------
// Per-head LayerNorm over D=64, f32 in place. One wave per head-row.
// ---------------------------------------------------------------------------
__global__ void __launch_bounds__(256)
headln_kernel(float* __restrict__ x, const float* __restrict__ g,
              const float* __restrict__ b, int rows) {
  int row = blockIdx.x * 8 + (threadIdx.x >> 5);
  int lane = threadIdx.x & 31;
  (void)rows;
  float* p = x + (size_t)row * 64;
  float a0 = p[lane], a1 = p[lane + 32];
  float s = a0 + a1;
#pragma unroll
  for (int m = 16; m >= 1; m >>= 1) s += __shfl_xor(s, m, 32);
  float mean = s * (1.f / 64.f);
  float d0 = a0 - mean, d1 = a1 - mean;
  float vv = d0 * d0 + d1 * d1;
#pragma unroll
  for (int m = 16; m >= 1; m >>= 1) vv += __shfl_xor(vv, m, 32);
  float rstd = rsqrtf(vv * (1.f / 64.f) + EPSF);
  p[lane]      = d0 * rstd * g[lane]      + b[lane];
  p[lane + 32] = d1 * rstd * g[lane + 32] + b[lane + 32];
}

// ---------------------------------------------------------------------------
// out = LN(a + alpha*b) over E=768. One wave per row, 24 elems/lane.
// ---------------------------------------------------------------------------
__global__ void __launch_bounds__(256)
addln_kernel(const float* __restrict__ a, const float* __restrict__ b,
             float alpha, const float* __restrict__ g,
             const float* __restrict__ be, float* __restrict__ out, int rows) {
  int row = blockIdx.x * 8 + (threadIdx.x >> 5);
  int lane = threadIdx.x & 31;
  (void)rows;
  const float* pa = a + (size_t)row * HDIM;
  const float* pb = b + (size_t)row * HDIM;
  float vals[24];
  float s = 0.f;
#pragma unroll
  for (int i = 0; i < 24; ++i) {
    int idx = lane + i * 32;
    float x = pa[idx] + alpha * pb[idx];
    vals[i] = x;
    s += x;
  }
#pragma unroll
  for (int m = 16; m >= 1; m >>= 1) s += __shfl_xor(s, m, 32);
  float mean = s * (1.f / 768.f);
  float vv = 0.f;
#pragma unroll
  for (int i = 0; i < 24; ++i) {
    float d = vals[i] - mean;
    vv += d * d;
  }
#pragma unroll
  for (int m = 16; m >= 1; m >>= 1) vv += __shfl_xor(vv, m, 32);
  float rstd = rsqrtf(vv * (1.f / 768.f) + EPSF);
  float* po = out + (size_t)row * HDIM;
#pragma unroll
  for (int i = 0; i < 24; ++i) {
    int idx = lane + i * 32;
    po[idx] = (vals[i] - mean) * rstd * g[idx] + be[idx];
  }
}

// ---------------------------------------------------------------------------
// Attention: one wave per 16 queries of one (b,h). q,k,v are f16 [rows, H*D],
// o is f32. Dynamic LDS: 16*Sk f32 scores.
// ---------------------------------------------------------------------------
__global__ void __launch_bounds__(32)
attn_kernel(const _Float16* __restrict__ q, const _Float16* __restrict__ k,
            const _Float16* __restrict__ v, float* __restrict__ o,
            int Sq, int Sk) {
  extern __shared__ float smem[];  // [16][Sk]
  __shared__ float rinv[16];
  const int lane = threadIdx.x;
  const int qbase = blockIdx.x * 16;
  const int h = blockIdx.y;
  const int b = blockIdx.z;
  const int hi = lane >> 4;
  const int lo = lane & 15;
  const _Float16 hs = (_Float16)SCALEF;

  // Q A-fragments (pre-scaled): two contiguous 16B vector loads per chunk.
  HFrag qa[2];
  {
    const _Float16* qrow =
        q + (size_t)(b * Sq + qbase + lo) * HDIM + h * DHEAD;
#pragma unroll
    for (int c = 0; c < 2; ++c) {
      const _Float16* p = qrow + c * 32 + (hi << 3);
      qa[c].h[0] = *(const v8h*)p * hs;
      qa[c].h[1] = *(const v8h*)(p + 16) * hs;
    }
  }

  // ---- scores via WMMA, 16-key blocks ----
  const int nkb = Sk >> 4;
  for (int kb = 0; kb < nkb; ++kb) {
    HFrag kf[2];
    const _Float16* krow =
        k + (size_t)(b * Sk + kb * 16 + lo) * HDIM + h * DHEAD;
#pragma unroll
    for (int c = 0; c < 2; ++c) {
      const _Float16* p = krow + c * 32 + (hi << 3);
      kf[c].h[0] = *(const v8h*)p;
      kf[c].h[1] = *(const v8h*)(p + 16);
    }
    v8f acc = vzero8();
    acc = wmma16(qa[0], kf[0], acc);
    acc = wmma16(qa[1], kf[1], acc);
    int key = kb * 16 + lo;
#pragma unroll
    for (int r = 0; r < 8; ++r)
      smem[(size_t)(r + (hi << 3)) * Sk + key] = acc[r];
  }
  __syncthreads();

  // ---- softmax per row (2 lanes per row) ----
  {
    int row = lane >> 1, part = lane & 1;
    float* srow = smem + (size_t)row * Sk;
    float m = -3.0e38f;
    for (int i = part; i < Sk; i += 2) m = fmaxf(m, srow[i]);
    m = fmaxf(m, __shfl_xor(m, 1, 32));
    float s = 0.f;
    for (int i = part; i < Sk; i += 2) {
      float e = __expf(srow[i] - m);
      srow[i] = e;
      s += e;
    }
    s += __shfl_xor(s, 1, 32);
    if (part == 0) rinv[row] = 1.f / s;
  }
  __syncthreads();

  // ---- O = P @ V via WMMA, 32-key blocks x 4 dim-tiles ----
  v8f oacc[4];
#pragma unroll
  for (int nc = 0; nc < 4; ++nc) oacc[nc] = vzero8();
  const float pr = rinv[lo];
  const int nkb2 = Sk >> 5;
  for (int kb = 0; kb < nkb2; ++kb) {
    HFrag pf;
#pragma unroll
    for (int e = 0; e < 16; ++e) {
      int key = kb * 32 + ((e < 8) ? e : e + 8) + (hi << 3);
      pf.v[e] = f2h(smem[(size_t)lo * Sk + key] * pr);
    }
#pragma unroll
    for (int nc = 0; nc < 4; ++nc) {
      HFrag vf;
#pragma unroll
      for (int e = 0; e < 16; ++e) {
        int key = kb * 32 + ((e < 8) ? e : e + 8) + (hi << 3);
        vf.v[e] = v[(size_t)(b * Sk + key) * HDIM + h * DHEAD + nc * 16 + lo];
      }
      oacc[nc] = wmma16(pf, vf, oacc[nc]);
    }
  }
#pragma unroll
  for (int nc = 0; nc < 4; ++nc) {
#pragma unroll
    for (int r = 0; r < 8; ++r) {
      int row = qbase + (hi << 3) + r;
      o[(size_t)(b * Sq + row) * HDIM + h * DHEAD + nc * 16 + lo] = oacc[nc][r];
    }
  }
}

// ---------------------------------------------------------------------------
extern "C" void kernel_launch(void* const* d_in, const int* in_sizes, int n_in,
                              void* d_out, int out_size, void* d_ws,
                              size_t ws_size, hipStream_t stream) {
  (void)in_sizes; (void)n_in; (void)out_size; (void)ws_size;
  const float* emb    = (const float*)d_in[0];
  const float* ctx    = (const float*)d_in[1];
  const float* sa_wq  = (const float*)d_in[2];
  const float* sa_wk  = (const float*)d_in[3];
  const float* sa_wv  = (const float*)d_in[4];
  const float* sa_wo  = (const float*)d_in[5];
  const float* sa_wob = (const float*)d_in[6];
  const float* sa_qng = (const float*)d_in[7];
  const float* sa_qnb = (const float*)d_in[8];
  const float* sa_kng = (const float*)d_in[9];
  const float* sa_knb = (const float*)d_in[10];
  const float* ca_wq  = (const float*)d_in[11];
  const float* ca_wk  = (const float*)d_in[12];
  const float* ca_wv  = (const float*)d_in[13];
  const float* ca_wo  = (const float*)d_in[14];
  const float* ca_wob = (const float*)d_in[15];
  const float* ca_qng = (const float*)d_in[16];
  const float* ca_qnb = (const float*)d_in[17];
  const float* ca_kng = (const float*)d_in[18];
  const float* ca_knb = (const float*)d_in[19];
  const float* mlp_w1 = (const float*)d_in[20];
  const float* mlp_b1 = (const float*)d_in[21];
  const float* mlp_w2 = (const float*)d_in[22];
  const float* mlp_b2 = (const float*)d_in[23];
  const float* ln1_g  = (const float*)d_in[24];
  const float* ln1_b  = (const float*)d_in[25];
  const float* ln2_g  = (const float*)d_in[26];
  const float* ln2_b  = (const float*)d_in[27];
  const float* ln3_g  = (const float*)d_in[28];
  const float* ln3_b  = (const float*)d_in[29];
  float* out = (float*)d_out;

  // ---- workspace carve-up ----
  const size_t N1  = (size_t)8192 * 768;   // 6291456
  const size_t NH1 = (size_t)8192 * 3072;  // 25165824
  float* t0 = (float*)d_ws;
  float* t1 = t0 + N1;
  float* t2 = t1 + N1;
  float* t3 = t2 + N1;
  float* th = t3 + N1;                       // 8192x3072 f32
  _Float16* hA = (_Float16*)(th + NH1);      // up to 8192x3072 f16
  _Float16* hQ = hA + NH1;
  _Float16* hK = hQ + N1;
  _Float16* hV = hK + N1;
  _Float16* hC = hV + N1;                    // 2048x512 f16
  _Float16* wt = hC + (size_t)2048 * 512;
  _Float16* wt_saq = wt;
  _Float16* wt_sak = wt_saq + 589824;
  _Float16* wt_sav = wt_sak + 589824;
  _Float16* wt_sao = wt_sav + 589824;
  _Float16* wt_caq = wt_sao + 589824;
  _Float16* wt_cak = wt_caq + 589824;   // [768][512]
  _Float16* wt_cav = wt_cak + 393216;   // [768][512]
  _Float16* wt_cao = wt_cav + 393216;
  _Float16* wt_m1  = wt_cao + 589824;   // [3072][768]
  _Float16* wt_m2  = wt_m1 + 2359296;   // [768][3072]

  const dim3 blk(256);

  // ---- one-time weight transpose+convert (f32 [K][N] -> f16 [N][K]) ----
  twcvt_kernel<<<dim3(24, 24), blk, 0, stream>>>(sa_wq, wt_saq, 768, 768);
  twcvt_kernel<<<dim3(24, 24), blk, 0, stream>>>(sa_wk, wt_sak, 768, 768);
  twcvt_kernel<<<dim3(24, 24), blk, 0, stream>>>(sa_wv, wt_sav, 768, 768);
  twcvt_kernel<<<dim3(24, 24), blk, 0, stream>>>(sa_wo, wt_sao, 768, 768);
  twcvt_kernel<<<dim3(24, 24), blk, 0, stream>>>(ca_wq, wt_caq, 768, 768);
  twcvt_kernel<<<dim3(24, 16), blk, 0, stream>>>(ca_wk, wt_cak, 512, 768);
  twcvt_kernel<<<dim3(24, 16), blk, 0, stream>>>(ca_wv, wt_cav, 512, 768);
  twcvt_kernel<<<dim3(24, 24), blk, 0, stream>>>(ca_wo, wt_cao, 768, 768);
  twcvt_kernel<<<dim3(96, 24), blk, 0, stream>>>(mlp_w1, wt_m1, 768, 3072);
  twcvt_kernel<<<dim3(24, 96), blk, 0, stream>>>(mlp_w2, wt_m2, 3072, 768);

  // ---- self-attention ----
  cvt_kernel<<<3072, blk, 0, stream>>>(emb, hA);  // emb f16
  gemm_h_kernel<0><<<dim3(6, 64), blk, 0, stream>>>(hA, wt_saq, nullptr, t0, 8192, 768, 768);
  gemm_h_kernel<0><<<dim3(6, 64), blk, 0, stream>>>(hA, wt_sak, nullptr, t1, 8192, 768, 768);
  gemm_h_kernel<0><<<dim3(6, 64), blk, 0, stream>>>(hA, wt_sav, nullptr, t2, 8192, 768, 768);
  headln_kernel<<<12288, blk, 0, stream>>>(t0, sa_qng, sa_qnb, 98304);
  headln_kernel<<<12288, blk, 0, stream>>>(t1, sa_kng, sa_knb, 98304);
  cvt_kernel<<<3072, blk, 0, stream>>>(t0, hQ);
  cvt_kernel<<<3072, blk, 0, stream>>>(t1, hK);
  cvt_kernel<<<3072, blk, 0, stream>>>(t2, hV);
  attn_kernel<<<dim3(64, 12, 8), 32, 16 * 1024 * sizeof(float), stream>>>(
      hQ, hK, hV, t3, 1024, 1024);
  cvt_kernel<<<3072, blk, 0, stream>>>(t3, hA);
  gemm_h_kernel<0><<<dim3(6, 64), blk, 0, stream>>>(hA, wt_sao, sa_wob, t0, 8192, 768, 768);
  addln_kernel<<<1024, blk, 0, stream>>>(t0, emb, 2.f, ln1_g, ln1_b, t1, 8192);

  // ---- cross-attention ----
  cvt_kernel<<<3072, blk, 0, stream>>>(t1, hA);   // x1 f16
  gemm_h_kernel<0><<<dim3(6, 64), blk, 0, stream>>>(hA, wt_caq, nullptr, t0, 8192, 768, 768);
  cvt_kernel<<<512, blk, 0, stream>>>(ctx, hC);
  gemm_h_kernel<0><<<dim3(6, 16), blk, 0, stream>>>(hC, wt_cak, nullptr, t2, 2048, 512, 768);
  gemm_h_kernel<0><<<dim3(6, 16), blk, 0, stream>>>(hC, wt_cav, nullptr, t3, 2048, 512, 768);
  headln_kernel<<<12288, blk, 0, stream>>>(t0, ca_qng, ca_qnb, 98304);
  headln_kernel<<<3072, blk, 0, stream>>>(t2, ca_kng, ca_knb, 24576);
  cvt_kernel<<<3072, blk, 0, stream>>>(t0, hQ);
  cvt_kernel<<<768, blk, 0, stream>>>(t2, hK);
  cvt_kernel<<<768, blk, 0, stream>>>(t3, hV);
  attn_kernel<<<dim3(64, 12, 8), 32, 16 * 256 * sizeof(float), stream>>>(
      hQ, hK, hV, th, 1024, 256);
  cvt_kernel<<<3072, blk, 0, stream>>>(th, hA);
  gemm_h_kernel<0><<<dim3(6, 64), blk, 0, stream>>>(hA, wt_cao, ca_wob, t0, 8192, 768, 768);
  addln_kernel<<<1024, blk, 0, stream>>>(t1, t0, 1.f, ln2_g, ln2_b, t2, 8192);

  // ---- MLP (exact GELU) + final LN ----
  cvt_kernel<<<3072, blk, 0, stream>>>(t2, hA);   // x2 f16
  gemm_h_kernel<1><<<dim3(24, 64), blk, 0, stream>>>(hA, wt_m1, mlp_b1, th, 8192, 768, 3072);
  cvt_kernel<<<12288, blk, 0, stream>>>(th, hA);  // gelu out f16
  gemm_h_kernel<0><<<dim3(6, 64), blk, 0, stream>>>(hA, wt_m2, mlp_b2, t0, 8192, 3072, 768);
  addln_kernel<<<1024, blk, 0, stream>>>(t0, t0, 0.f, ln3_g, ln3_b, out, 8192);
}
